// FusedODModel_83837761618206
// MI455X (gfx1250) — matrix operand order repaired and checked by reading, not compiled, not run on hardware
//
#include <hip/hip_runtime.h>
#include <hip/hip_bf16.h>

typedef __attribute__((ext_vector_type(16))) _Float16 v16h;
typedef __attribute__((ext_vector_type(8)))  _Float16 v8h;
typedef __attribute__((ext_vector_type(4)))  _Float16 v4h;
typedef __attribute__((ext_vector_type(8)))  float    v8f;

#define NN   256
#define HID  64
#define TT   144
#define BB   4
#define CHS  9
#define NCH  4

__device__ __forceinline__ float sigf(float x) { return 1.0f / (1.0f + __expf(-x)); }

// ---- WMMA fragment helpers (wave32, 16x16x32 f16 -> f32) -------------------
// A: 16x32 tile from row-major f16 LDS (ld in halves).
__device__ __forceinline__ v16h load_frag_a(const _Float16* base, int ld, int m0, int k0, int lane) {
    int r   = m0 + (lane & 15);
    int off = k0 + ((lane >> 4) << 4);   // lanes 16-31 take K+16
    const _Float16* p = base + r * ld + off;
    v8h lo = *(const v8h*)(p);           // K = off .. off+7
    v8h hi = *(const v8h*)(p + 8);       // K = off+8 .. off+15
    return __builtin_shufflevector(lo, hi, 0,1,2,3,4,5,6,7,8,9,10,11,12,13,14,15);
}
// B: 32x16 tile from K-contiguous (transposed) f16 LDS: elem(k,n) at baseT[n*ldt + k].
__device__ __forceinline__ v16h load_frag_b(const _Float16* baseT, int ldt, int n0, int k0, int lane) {
    int n  = n0 + (lane & 15);
    int ko = k0 + ((lane >> 4) << 4);    // lanes 16-31 take K+16
    return *(const v16h*)(baseT + n * ldt + ko);
}
// C/D: lane = column, VGPR v = row (rows m0.. / m0+8.. for upper lanes).
__device__ __forceinline__ void store_frag_c(float* dst, int ld, int m0, int n0, int lane, v8f c) {
    int col = n0 + (lane & 15);
    int rb  = m0 + ((lane >> 4) << 3);
#pragma unroll
    for (int v = 0; v < 8; ++v) dst[(rb + v) * ld + col] = c[v];
}

// ---------------------------------------------------------------------------
// Kernel 1: fused GAT per (b,t): Wh = seg@Wg (WMMA), masked softmax, Hn = attn@Wh
// (WMMA), elu, mean over nodes -> feats[b,t,:]
// ---------------------------------------------------------------------------
extern __shared__ char smem[];

__global__ __launch_bounds__(256) void gat_kernel(
    const float* __restrict__ seg, const float* __restrict__ adj,
    const float* __restrict__ dist, const float* __restrict__ Wg,
    const float* __restrict__ a_src, const float* __restrict__ a_dst,
    float* __restrict__ feats)
{
    _Float16* sA   = (_Float16*)(smem);            // 256x256 f16 : seg, later attn (128 KB)
    _Float16* sBT  = (_Float16*)(smem + 131072);   // 64x256  f16 : WgT, later WhT   (32 KB)
    float*    sWh  = (float*)   (smem + 163840);   // 256x64  f32                    (64 KB)
    float*    sSrc = (float*)   (smem + 229376);   // 256
    float*    sDst = (float*)   (smem + 230400);   // 256
    float*    sVs  = (float*)   (smem + 231424);   // 64
    float*    sVd  = (float*)   (smem + 231680);   // 64
    float*    sFeat= (float*)   (smem + 231936);   // 64

    const int tid  = threadIdx.x;
    const int lane = tid & 31;
    // wave id is uniform across the wave: force it scalar so tile loops compile
    // to SALU loops (EXEC stays all-1s around every v_wmma, per ISA requirement).
    const int wave = __builtin_amdgcn_readfirstlane(tid >> 5);
    const int bt   = blockIdx.x;
    const float* segbt = seg + (size_t)bt * (NN * NN);

    // ---- stage: seg -> f16, W_gat -> transposed f16, attention vectors ----
    {
        const float4* s4 = (const float4*)segbt;
        for (int i = tid; i < NN * NN / 4; i += 256) {
            __builtin_prefetch(&s4[i + 1024], 0, 1);
            float4 v = s4[i];
            v4h h; h[0] = (_Float16)v.x; h[1] = (_Float16)v.y;
                   h[2] = (_Float16)v.z; h[3] = (_Float16)v.w;
            *(v4h*)(sA + 4 * i) = h;
        }
    }
    for (int i = tid; i < NN * HID; i += 256) {
        int f = i >> 6, h = i & 63;
        sBT[h * NN + f] = (_Float16)Wg[i];
    }
    if (tid < 64) { sVs[tid] = a_src[tid]; sVd[tid] = a_dst[tid]; }
    __syncthreads();

    // ---- GEMM1: Wh = seg(256x256) @ Wg(256x64) ---------------------------
    // wave w owns m-tiles 2w..2w+1 x all 4 n-tiles; uniform trip count.
    for (int i = 0; i < 8; ++i) {
        int t8 = wave * 8 + i;
        int m0 = (t8 >> 2) << 4;
        int n0 = (t8 & 3) << 4;
        v8f c = {};
#pragma unroll
        for (int k0 = 0; k0 < 256; k0 += 32) {
            v16h a = load_frag_a(sA,  NN, m0, k0, lane);
            v16h b = load_frag_b(sBT, NN, n0, k0, lane);
            c = __builtin_amdgcn_wmma_f32_16x16x32_f16(false, a, false, b, (short)0, c, false, false);
        }
        store_frag_c(sWh, HID, m0, n0, lane, c);
    }
    __syncthreads();

    // ---- per-row scores + transpose Wh -> f16 (overwrites WgT) -----------
    {
        float aS = 0.f, aD = 0.f;
        for (int h = 0; h < 64; ++h) {
            float w = sWh[tid * 64 + h];
            aS += w * sVs[h]; aD += w * sVd[h];
        }
        sSrc[tid] = aS; sDst[tid] = aD;
    }
    for (int i = tid; i < NN * HID; i += 256) {
        int r = i >> 6, h = i & 63;
        sBT[h * NN + r] = (_Float16)sWh[i];
    }
    if (tid < 64) sFeat[tid] = 0.f;
    __syncthreads();

    // ---- masked softmax; attn (f16) overwrites seg tile -------------------
    for (int i = 0; i < 32; ++i) {
        int r = wave * 32 + i;
        float li[8];
        float m = -3.4e38f;
        float ssrc = sSrc[r];
#pragma unroll
        for (int jj = 0; jj < 8; ++jj) {
            int j = lane + jj * 32;
            float e = ssrc + sDst[j];
            e = (e >= 0.f) ? e : 0.2f * e;              // leaky_relu 0.2
            e -= dist[r * NN + j];
            e = (adj[r * NN + j] > 0.f) ? e : -1e9f;
            li[jj] = e;
            m = fmaxf(m, e);
        }
#pragma unroll
        for (int o = 16; o >= 1; o >>= 1) m = fmaxf(m, __shfl_xor(m, o, 32));
        float s = 0.f;
#pragma unroll
        for (int jj = 0; jj < 8; ++jj) { float p = __expf(li[jj] - m); li[jj] = p; s += p; }
#pragma unroll
        for (int o = 16; o >= 1; o >>= 1) s += __shfl_xor(s, o, 32);
        float inv = 1.0f / s;
#pragma unroll
        for (int jj = 0; jj < 8; ++jj)
            sA[r * NN + lane + jj * 32] = (_Float16)(li[jj] * inv);
    }
    __syncthreads();

    // ---- GEMM2: Hn = attn(256x256) @ Wh(256x64); elu; column sums ---------
    for (int i = 0; i < 8; ++i) {
        int t8 = wave * 8 + i;
        int m0 = (t8 >> 2) << 4;
        int n0 = (t8 & 3) << 4;
        v8f c = {};
#pragma unroll
        for (int k0 = 0; k0 < 256; k0 += 32) {
            v16h a = load_frag_a(sA,  NN, m0, k0, lane);
            v16h b = load_frag_b(sBT, NN, n0, k0, lane);
            c = __builtin_amdgcn_wmma_f32_16x16x32_f16(false, a, false, b, (short)0, c, false, false);
        }
        int col = n0 + (lane & 15);
        float part = 0.f;
#pragma unroll
        for (int v = 0; v < 8; ++v) {
            float cv = c[v];
            part += (cv > 0.f) ? cv : (__expf(cv) - 1.0f);   // elu
        }
        atomicAdd(&sFeat[col], part);                         // ds_add_f32
    }
    __syncthreads();

    if (tid < 64) feats[bt * HID + tid] = sFeat[tid] * (1.0f / 256.0f);
}

// ---------------------------------------------------------------------------
// Kernel 2: LSTM + tattn_L + 4x(GRU + tattn_S) + lin1  (single persistent block)
// ---------------------------------------------------------------------------
__global__ __launch_bounds__(256) void seq_kernel(
    const float* __restrict__ feats,
    const float* __restrict__ lWih, const float* __restrict__ lWhh,
    const float* __restrict__ lbih, const float* __restrict__ lbhh,
    const float* __restrict__ aLW,  const float* __restrict__ aLb, const float* __restrict__ aLv,
    const float* __restrict__ gWih, const float* __restrict__ gWhh,
    const float* __restrict__ gbih, const float* __restrict__ gbhh,
    const float* __restrict__ aSW,  const float* __restrict__ aSb, const float* __restrict__ aSv,
    const float* __restrict__ l1W,  const float* __restrict__ l1b,
    float* __restrict__ h1out)
{
    float* ys  = (float*)(smem);             // [4][144][64]  147456 B
    float* gbx = (float*)(smem + 147456);    // 1024 floats (gates / gru x-gates)
    float* gbh = (float*)(smem + 151552);    // 1024 floats (gru h-gates)
    float* hb  = (float*)(smem + 155648);    // [4][64]
    float* cb  = (float*)(smem + 156672);    // [4][64]
    float* sc  = (float*)(smem + 157696);    // 576 scores
    float* ctx = (float*)(smem + 160000);    // [4][320] fused
    float* gys = (float*)(smem + 165120);    // [4][9][64] per-chunk GRU outputs
    float* gh  = (float*)(smem + 174336);    // [4][64]

    const int tid = threadIdx.x;

    // ---------------- LSTM over T=144 ----------------
    hb[tid] = 0.f; cb[tid] = 0.f;
    __syncthreads();
    for (int t = 0; t < TT; ++t) {
#pragma unroll
        for (int q = 0; q < 4; ++q) {
            int g = tid + q * 256;               // 1024 gate elems: [b][256]
            int b = g >> 8, row = g & 255;
            const float* wi = lWih + row * 64;
            const float* wh = lWhh + row * 64;
            const float* x  = feats + (b * TT + t) * 64;
            const float* h  = hb + b * 64;
            float acc = lbih[row] + lbhh[row];
            for (int k = 0; k < 64; ++k) acc += wi[k] * x[k] + wh[k] * h[k];
            gbx[g] = acc;
        }
        __syncthreads();
        {
            int b = tid >> 6, j = tid & 63;
            float gi = gbx[b * 256 + j];
            float gf = gbx[b * 256 + 64 + j];
            float gg = gbx[b * 256 + 128 + j];
            float go = gbx[b * 256 + 192 + j];
            float c  = sigf(gf) * cb[tid] + sigf(gi) * tanhf(gg);
            float h  = sigf(go) * tanhf(c);
            cb[tid] = c; hb[tid] = h;
            ys[(b * TT + t) * 64 + j] = h;
        }
        __syncthreads();
    }

    // ---------------- temporal attention (long) ----------------
    for (int p = tid; p < BB * TT; p += 256) {
        int b = p / TT, t = p % TT;
        const float* h = ys + (b * TT + t) * 64;
        float s = 0.f;
        for (int ho = 0; ho < 64; ++ho) {
            float acc = aLb[ho];
            for (int k = 0; k < 64; ++k) acc += h[k] * aLW[k * 64 + ho];
            s += aLv[ho] * tanhf(acc);
        }
        sc[p] = s;
    }
    __syncthreads();
    if (tid < BB) {
        float m = -3.4e38f;
        for (int t = 0; t < TT; ++t) m = fmaxf(m, sc[tid * TT + t]);
        float sum = 0.f;
        for (int t = 0; t < TT; ++t) { float p = __expf(sc[tid * TT + t] - m); sc[tid * TT + t] = p; sum += p; }
        float inv = 1.0f / sum;
        for (int t = 0; t < TT; ++t) sc[tid * TT + t] *= inv;
    }
    __syncthreads();
    {
        int b = tid >> 6, j = tid & 63;
        float acc = 0.f;
        for (int t = 0; t < TT; ++t) acc += sc[b * TT + t] * ys[(b * TT + t) * 64 + j];
        ctx[b * 320 + j] = acc;
    }
    __syncthreads();

    // ---------------- 4 GRU chunks + short attention ----------------
    for (int c = 0; c < NCH; ++c) {
        gh[tid] = 0.f;
        __syncthreads();
        for (int st = 0; st < CHS; ++st) {
            for (int q = tid; q < BB * 192; q += 256) {
                int b = q / 192, row = q % 192;
                const float* wi = gWih + (c * 192 + row) * 64;
                const float* wh = gWhh + (c * 192 + row) * 64;
                const float* x  = feats + (b * TT + (TT - CHS * NCH) + c * CHS + st) * 64;
                const float* h  = gh + b * 64;
                float ax = gbih[c * 192 + row], ah = gbhh[c * 192 + row];
                for (int k = 0; k < 64; ++k) { ax += wi[k] * x[k]; ah += wh[k] * h[k]; }
                gbx[q] = ax; gbh[q] = ah;
            }
            __syncthreads();
            {
                int b = tid >> 6, j = tid & 63;
                float xr = gbx[b * 192 + j],       hr = gbh[b * 192 + j];
                float xz = gbx[b * 192 + 64 + j],  hz = gbh[b * 192 + 64 + j];
                float xn = gbx[b * 192 + 128 + j], hn = gbh[b * 192 + 128 + j];
                float rr = sigf(xr + hr), zz = sigf(xz + hz);
                float nn = tanhf(xn + rr * hn);
                float h  = (1.0f - zz) * nn + zz * gh[tid];
                gh[tid] = h;
                gys[(b * CHS + st) * 64 + j] = h;
            }
            __syncthreads();
        }
        for (int p = tid; p < BB * CHS; p += 256) {
            int b = p / CHS, t = p % CHS;
            const float* h = gys + (b * CHS + t) * 64;
            float s = 0.f;
            for (int ho = 0; ho < 64; ++ho) {
                float acc = aSb[c * 64 + ho];
                for (int k = 0; k < 64; ++k) acc += h[k] * aSW[(c * 64 + k) * 64 + ho];
                s += aSv[c * 64 + ho] * tanhf(acc);
            }
            sc[p] = s;
        }
        __syncthreads();
        if (tid < BB) {
            float m = -3.4e38f;
            for (int t = 0; t < CHS; ++t) m = fmaxf(m, sc[tid * CHS + t]);
            float sum = 0.f;
            for (int t = 0; t < CHS; ++t) { float p = __expf(sc[tid * CHS + t] - m); sc[tid * CHS + t] = p; sum += p; }
            float inv = 1.0f / sum;
            for (int t = 0; t < CHS; ++t) sc[tid * CHS + t] *= inv;
        }
        __syncthreads();
        {
            int b = tid >> 6, j = tid & 63;
            float acc = 0.f;
            for (int t = 0; t < CHS; ++t) acc += sc[b * CHS + t] * gys[(b * CHS + t) * 64 + j];
            ctx[b * 320 + 64 + c * 64 + j] = acc;
        }
        __syncthreads();
    }

    // ---------------- lin1 + relu ----------------
    for (int o = tid; o < BB * 128; o += 256) {
        int b = o >> 7, r = o & 127;
        float acc = l1b[r];
        const float* w = l1W + r * 320;
        const float* f = ctx + b * 320;
        for (int k = 0; k < 320; ++k) acc += w[k] * f[k];
        h1out[b * 128 + r] = fmaxf(acc, 0.f);
    }
}

// ---------------------------------------------------------------------------
// Kernel 3: out_pre[b,o] = h1[b,:] @ lin2_W[o,:] + lin2_b[o]   (bandwidth bound)
// ---------------------------------------------------------------------------
__global__ __launch_bounds__(256) void lin2_kernel(
    const float* __restrict__ h1, const float* __restrict__ W,
    const float* __restrict__ bias, float* __restrict__ outp)
{
    __shared__ float sh1[512];
    int tid = threadIdx.x;
    for (int i = tid; i < 512; i += 256) sh1[i] = h1[i];
    __syncthreads();
    int o = blockIdx.x * 256 + tid;
    const float* w = W + (size_t)o * 128;
    float bv = bias[o];
    float a0 = bv, a1 = bv, a2 = bv, a3 = bv;
    for (int k = 0; k < 128; ++k) {
        float wv = w[k];
        a0 += wv * sh1[k];
        a1 += wv * sh1[128 + k];
        a2 += wv * sh1[256 + k];
        a3 += wv * sh1[384 + k];
    }
    outp[o]          = a0;
    outp[65536 + o]  = a1;
    outp[131072 + o] = a2;
    outp[196608 + o] = a3;
}

// ---------------------------------------------------------------------------
// Kernel 4: 3x3 SAME conv (1 channel) + bias + softplus
// ---------------------------------------------------------------------------
__global__ __launch_bounds__(256) void conv_kernel(
    const float* __restrict__ inp, const float* __restrict__ k9,
    const float* __restrict__ kb, float* __restrict__ out)
{
    int idx = blockIdx.x * 256 + threadIdx.x;   // 0 .. 262143
    int b = idx >> 16, rem = idx & 65535;
    int y = rem >> 8, x = rem & 255;
    float acc = kb[0];
#pragma unroll
    for (int dy = 0; dy < 3; ++dy) {
        int yy = y + dy - 1;
        if (yy < 0 || yy > 255) continue;
#pragma unroll
        for (int dx = 0; dx < 3; ++dx) {
            int xx = x + dx - 1;
            if (xx < 0 || xx > 255) continue;
            acc += inp[(b << 16) + (yy << 8) + xx] * k9[dy * 3 + dx];
        }
    }
    out[idx] = (acc > 20.f) ? acc : log1pf(__expf(acc));
}

// ---------------------------------------------------------------------------
extern "C" void kernel_launch(void* const* d_in, const int* in_sizes, int n_in,
                              void* d_out, int out_size, void* d_ws, size_t ws_size,
                              hipStream_t stream)
{
    const float* seg      = (const float*)d_in[0];
    const float* adj      = (const float*)d_in[1];
    const float* dist     = (const float*)d_in[2];
    const float* W_gat    = (const float*)d_in[3];
    const float* a_src    = (const float*)d_in[4];
    const float* a_dst    = (const float*)d_in[5];
    const float* lstm_Wih = (const float*)d_in[6];
    const float* lstm_Whh = (const float*)d_in[7];
    const float* lstm_bih = (const float*)d_in[8];
    const float* lstm_bhh = (const float*)d_in[9];
    const float* attnL_W  = (const float*)d_in[10];
    const float* attnL_b  = (const float*)d_in[11];
    const float* attnL_v  = (const float*)d_in[12];
    const float* gru_Wih  = (const float*)d_in[13];
    const float* gru_Whh  = (const float*)d_in[14];
    const float* gru_bih  = (const float*)d_in[15];
    const float* gru_bhh  = (const float*)d_in[16];
    const float* attnS_W  = (const float*)d_in[17];
    const float* attnS_b  = (const float*)d_in[18];
    const float* attnS_v  = (const float*)d_in[19];
    const float* lin1_W   = (const float*)d_in[20];
    const float* lin1_b   = (const float*)d_in[21];
    const float* lin2_W   = (const float*)d_in[22];
    const float* lin2_b   = (const float*)d_in[23];
    const float* conv_k   = (const float*)d_in[24];
    const float* conv_b   = (const float*)d_in[25];

    float* wsFeats = (float*)d_ws;                          // 4*144*64 f32 = 147456 B
    float* wsH1    = (float*)((char*)d_ws + 147456);        // 4*128 f32    = 2048 B
    float* wsPre   = (float*)((char*)d_ws + 149504);        // 4*65536 f32  = 1 MB

    const int smem1 = 232192;   // 227 KB, fits the 320 KB WGP LDS
    const int smem2 = 175360;
    hipFuncSetAttribute(reinterpret_cast<const void*>(gat_kernel),
                        hipFuncAttributeMaxDynamicSharedMemorySize, smem1);
    hipFuncSetAttribute(reinterpret_cast<const void*>(seq_kernel),
                        hipFuncAttributeMaxDynamicSharedMemorySize, smem2);

    gat_kernel<<<BB * TT, 256, smem1, stream>>>(seg, adj, dist, W_gat, a_src, a_dst, wsFeats);

    seq_kernel<<<1, 256, smem2, stream>>>(wsFeats,
        lstm_Wih, lstm_Whh, lstm_bih, lstm_bhh,
        attnL_W, attnL_b, attnL_v,
        gru_Wih, gru_Whh, gru_bih, gru_bhh,
        attnS_W, attnS_b, attnS_v,
        lin1_W, lin1_b, wsH1);

    lin2_kernel<<<256, 256, 0, stream>>>(wsH1, lin2_W, lin2_b, wsPre);

    conv_kernel<<<1024, 256, 0, stream>>>(wsPre, conv_k, conv_b, (float*)d_out);
}
